// GraphTemporalLayer_66752381715031
// MI455X (gfx1250) — matrix operand order repaired and checked by reading, not compiled, not run on hardware
//
#include <hip/hip_runtime.h>
#include <hip/hip_bf16.h>

// ---------------------------------------------------------------------------
// GraphTemporalLayer for MI455X (gfx1250, wave32, WMMA)
// B=4096, T=16, H=512, NH=16, HD=32
// ---------------------------------------------------------------------------

#define Tn 16
#define Hn 512
#define NHn 16
#define HDn 32
#define XS 520              // f16 row stride (pad 8 halves -> conflict-free b128 DS loads)
#define FS 516              // f32 row stride (pad 4 floats)
#define EPSv 1e-5f

typedef __attribute__((ext_vector_type(16))) _Float16 v16h;
typedef __attribute__((ext_vector_type(8)))  float    v8f;

union FragH {
    v16h  h;
    uint4 u[2];
};

// ---------------------------------------------------------------------------
// Prep kernel: convert Wq,Wk,Wv,ff1_w (f32, [k][n]) -> transposed f16 [n][k]
// so each lane's WMMA B-fragment is two contiguous 16-byte k runs.
// ---------------------------------------------------------------------------
__global__ __launch_bounds__(256) void prep_weights_kernel(
    const float* __restrict__ wq, const float* __restrict__ wk,
    const float* __restrict__ wv, const float* __restrict__ ff1,
    _Float16* __restrict__ outw)
{
    int idx = blockIdx.x * 256 + threadIdx.x;      // 4 * 512 * 512 total
    int m = idx >> 18;                             // which matrix
    int r = idx & 262143;
    int k = r & 511;                               // contiguous k per thread -> coalesced store
    int n = r >> 9;
    const float* src = (m == 0) ? wq : (m == 1) ? wk : (m == 2) ? wv : ff1;
    outw[(size_t)m * 262144 + (size_t)n * 512 + k] = (_Float16)src[(size_t)k * 512 + n];
}

// ---------------------------------------------------------------------------
// Main fused kernel: one workgroup per batch element b.
// ---------------------------------------------------------------------------
__global__ __launch_bounds__(256) void graph_temporal_kernel(
    const float* __restrict__ structural, const float* __restrict__ pos_cc,
    const float* __restrict__ pos_blr,    const float* __restrict__ pe,
    const float* __restrict__ gamma,      const float* __restrict__ beta,
    const float* __restrict__ ff1_b,      const float* __restrict__ prelu_a,
    const float* __restrict__ ff2_w,      const float* __restrict__ ff2_b,
    const _Float16* __restrict__ wT,      float* __restrict__ out)
{
    extern __shared__ char smem[];
    _Float16* xln     = (_Float16*)smem;                 // [16][XS]  LN'd x (reused for LN2 out)
    _Float16* qkv     = xln + 16 * XS;                   // [3][16][XS]
    float*    sstruct = (float*)(smem + 66560);          // [16][FS]  structural (for residual)
    float*    attnw   = sstruct + 16 * FS;               // [8][16][17] per-wave attn weights
    float*    hbuf    = attnw + 8 * 16 * 17;             // [16][FS]  attn-out, then FF1 out

    const int b    = blockIdx.x;
    const int tid  = threadIdx.x;
    const int lane = tid & 31;
    const int w    = tid >> 5;                           // wave 0..7
    const size_t base = (size_t)b * Tn * Hn;

    const int mlo  = lane & 15;
    const int moff = (lane & 16) ? 8 : 0;                // C-fragment row offset
    const int aseg = (lane & 16) ? 8 : 0;                // A-fragment K seg (halves)
    const int bseg = (lane & 16) ? 8 : 0;                // B-fragment K seg (halves)

    // ---------------- Stage A: x = s + pe + cc + blr ; LayerNorm -> f16 ----
    #pragma unroll
    for (int rr = 0; rr < 2; ++rr) {
        const int t = w * 2 + rr;
        float xv[16];
        float s1 = 0.f, s2 = 0.f;
        #pragma unroll
        for (int i = 0; i < 16; ++i) {
            int k = i * 32 + lane;
            float sv = structural[base + t * Hn + k];
            float v  = sv + pos_cc[base + t * Hn + k] + pos_blr[base + t * Hn + k]
                          + pe[t * Hn + k];
            sstruct[t * FS + k] = sv;
            xv[i] = v; s1 += v; s2 += v * v;
        }
        #pragma unroll
        for (int s = 1; s < 32; s <<= 1) { s1 += __shfl_xor(s1, s, 32); s2 += __shfl_xor(s2, s, 32); }
        float mu = s1 * (1.0f / Hn);
        float rs = rsqrtf(s2 * (1.0f / Hn) - mu * mu + EPSv);
        #pragma unroll
        for (int i = 0; i < 16; ++i) {
            int k = i * 32 + lane;
            xln[t * XS + k] = (_Float16)((xv[i] - mu) * rs * gamma[k] + beta[k]);
        }
    }
    __syncthreads();

    // ---------------- Stage B: Q,K,V = xln @ W  (WMMA f16 -> f32) ----------
    for (int g = 0; g < 3; ++g) {
        const _Float16* wmat = wT + (size_t)g * Hn * Hn;   // [n][k] f16
        #pragma unroll
        for (int nt = 0; nt < 4; ++nt) {
            const int n0 = (w * 4 + nt) * 16;
            const int n  = n0 + mlo;
            v8f c = {0.f, 0.f, 0.f, 0.f, 0.f, 0.f, 0.f, 0.f};
            for (int k0 = 0; k0 < Hn; k0 += 32) {
                FragH a, bb;
                a.u[0]  = *(const uint4*)&xln[mlo * XS + k0 + aseg];
                a.u[1]  = *(const uint4*)&xln[mlo * XS + k0 + aseg + 16];
                const uint4* bp = (const uint4*)&wmat[(size_t)n * Hn + k0 + bseg];
                bb.u[0] = bp[0];
                bb.u[1] = bp[2];
                c = __builtin_amdgcn_wmma_f32_16x16x32_f16(
                        false, a.h, false, bb.h, (short)0, c, false, false);
            }
            #pragma unroll
            for (int r = 0; r < 8; ++r)
                qkv[g * 16 * XS + (r + moff) * XS + n] = (_Float16)c[r];
        }
    }
    __syncthreads();

    // ---------------- Stage C: causal attention, 2 heads per wave ----------
    const float iscale = 0.17677669529663689f;   // 1/sqrt(32)
    #pragma unroll
    for (int hh = 0; hh < 2; ++hh) {
        const int hd = (w + hh * 8) * HDn;       // head column base
        FragH a, bb;
        // A = Q_head [16 x 32]
        a.u[0]  = *(const uint4*)&qkv[0 * 16 * XS + mlo * XS + hd + aseg];
        a.u[1]  = *(const uint4*)&qkv[0 * 16 * XS + mlo * XS + hd + aseg + 16];
        // B = K_head^T : B[d][tk] = K[tk][d]; lane holds col tk=mlo, K-seg bseg
        const uint4* bp = (const uint4*)&qkv[1 * 16 * XS + mlo * XS + hd + bseg];
        bb.u[0] = bp[0];
        bb.u[1] = bp[2];
        v8f c = {0.f, 0.f, 0.f, 0.f, 0.f, 0.f, 0.f, 0.f};
        c = __builtin_amdgcn_wmma_f32_16x16x32_f16(
                false, a.h, false, bb.h, (short)0, c, false, false);

        // mask + softmax along tk (16-lane groups of the C fragment)
        const int tk = mlo;
        #pragma unroll
        for (int r = 0; r < 8; ++r) {
            int mq  = r + moff;
            float v = (tk <= mq) ? c[r] * iscale : -3.0e38f;
            float mx = v;
            #pragma unroll
            for (int s = 1; s < 16; s <<= 1) mx = fmaxf(mx, __shfl_xor(mx, s, 32));
            float p = __expf(v - mx);
            float sum = p;
            #pragma unroll
            for (int s = 1; s < 16; s <<= 1) sum += __shfl_xor(sum, s, 32);
            attnw[(w * 16 + mq) * 17 + tk] = p / sum;
        }
        __syncthreads();

        // out_head[mq][d] = sum_tk attn[mq][tk] * V[tk][d] ; d = lane (0..31)
        for (int mq = 0; mq < 16; ++mq) {
            float acc = 0.f;
            #pragma unroll
            for (int t2 = 0; t2 < 16; ++t2)
                acc += attnw[(w * 16 + mq) * 17 + t2]
                     * (float)qkv[2 * 16 * XS + t2 * XS + hd + lane];
            hbuf[mq * FS + hd + lane] = acc;
        }
        __syncthreads();
    }

    // ---------------- Stage D: residual + LayerNorm2 -> xln (f16) ----------
    #pragma unroll
    for (int rr = 0; rr < 2; ++rr) {
        const int t = w * 2 + rr;
        float xv[16];
        float s1 = 0.f, s2 = 0.f;
        #pragma unroll
        for (int i = 0; i < 16; ++i) {
            int k = i * 32 + lane;
            float v = hbuf[t * FS + k] + sstruct[t * FS + k];
            xv[i] = v; s1 += v; s2 += v * v;
        }
        #pragma unroll
        for (int s = 1; s < 32; s <<= 1) { s1 += __shfl_xor(s1, s, 32); s2 += __shfl_xor(s2, s, 32); }
        float mu = s1 * (1.0f / Hn);
        float rs = rsqrtf(s2 * (1.0f / Hn) - mu * mu + EPSv);
        #pragma unroll
        for (int i = 0; i < 16; ++i) {
            int k = i * 32 + lane;
            xln[t * XS + k] = (_Float16)((xv[i] - mu) * rs * gamma[k] + beta[k]);
        }
    }
    __syncthreads();

    // ---------------- Stage E: FF1 (WMMA) + bias + PReLU -> hbuf (f32) -----
    {
        const float pa = prelu_a[0];
        const _Float16* wmat = wT + (size_t)3 * Hn * Hn;
        #pragma unroll
        for (int nt = 0; nt < 4; ++nt) {
            const int n0 = (w * 4 + nt) * 16;
            const int n  = n0 + mlo;
            v8f c = {0.f, 0.f, 0.f, 0.f, 0.f, 0.f, 0.f, 0.f};
            for (int k0 = 0; k0 < Hn; k0 += 32) {
                FragH a, bb;
                a.u[0]  = *(const uint4*)&xln[mlo * XS + k0 + aseg];
                a.u[1]  = *(const uint4*)&xln[mlo * XS + k0 + aseg + 16];
                const uint4* bp = (const uint4*)&wmat[(size_t)n * Hn + k0 + bseg];
                bb.u[0] = bp[0];
                bb.u[1] = bp[2];
                c = __builtin_amdgcn_wmma_f32_16x16x32_f16(
                        false, a.h, false, bb.h, (short)0, c, false, false);
            }
            const float bias = ff1_b[n];
            #pragma unroll
            for (int r = 0; r < 8; ++r) {
                float hv = c[r] + bias;
                hv = hv > 0.f ? hv : pa * hv;
                hbuf[(r + moff) * FS + n] = hv;
            }
        }
    }
    __syncthreads();

    // ---------------- Stage F: FF2 [16,512]@[512,2] + bias -----------------
    #pragma unroll
    for (int p = 0; p < 4; ++p) {
        const int idx = w * 4 + p;              // 0..31
        const int t = idx >> 1, j = idx & 1;
        float acc = 0.f;
        #pragma unroll
        for (int i = 0; i < 16; ++i) {
            int k = i * 32 + lane;
            acc += hbuf[t * FS + k] * ff2_w[k * 2 + j];
        }
        #pragma unroll
        for (int s = 1; s < 32; s <<= 1) acc += __shfl_xor(acc, s, 32);
        if (lane == 0) out[(size_t)b * 32 + t * 2 + j] = acc + ff2_b[j];
    }
}

// ---------------------------------------------------------------------------
extern "C" void kernel_launch(void* const* d_in, const int* in_sizes, int n_in,
                              void* d_out, int out_size, void* d_ws, size_t ws_size,
                              hipStream_t stream) {
    const float* structural = (const float*)d_in[0];
    const float* pos_cc     = (const float*)d_in[1];
    const float* pos_blr    = (const float*)d_in[2];
    // d_in[3] = exist_nodes (unused by reference)
    const float* Wq         = (const float*)d_in[4];
    const float* Wk         = (const float*)d_in[5];
    const float* Wv         = (const float*)d_in[6];
    const float* pe_table   = (const float*)d_in[7];
    const float* ln_gamma   = (const float*)d_in[8];
    const float* ln_beta    = (const float*)d_in[9];
    const float* ff1_w      = (const float*)d_in[10];
    const float* ff1_b      = (const float*)d_in[11];
    const float* prelu_a    = (const float*)d_in[12];
    const float* ff2_w      = (const float*)d_in[13];
    const float* ff2_b      = (const float*)d_in[14];
    float* outp             = (float*)d_out;

    const int Bdim = in_sizes[0] / (Tn * Hn);            // 4096

    _Float16* wT = (_Float16*)d_ws;                      // 4 * 512*512 f16 = 2 MB

    // one-time (per call) weight convert+transpose: 4*512*512 elements
    prep_weights_kernel<<<(4 * Hn * Hn) / 256, 256, 0, stream>>>(Wq, Wk, Wv, ff1_w, wT);

    const size_t smem_bytes = 141312;                    // see carve-out above
    graph_temporal_kernel<<<Bdim, 256, smem_bytes, stream>>>(
        structural, pos_cc, pos_blr, pe_table, ln_gamma, ln_beta,
        ff1_b, prelu_a, ff2_w, ff2_b, wT, outp);
}
